// QuakeMamba2_14482629722586
// MI455X (gfx1250) — matrix operand order repaired and checked by reading, not compiled, not run on hardware
//
#include <hip/hip_runtime.h>
#include <hip/hip_bf16.h>

// ---------------------------------------------------------------------------
// Mamba2-style classifier forward for MI455X (gfx1250, wave32, WMMA).
// T=8192 tokens. Big GEMMs: bf16 WMMA (f32 accumulate) with pre-converted
// bf16 operands, double-buffered LDS tiles staged via async-to-LDS DMA.
// Only the last token feeds out_proj/classifier (z computed for 8 rows only).
// ---------------------------------------------------------------------------

#define T_TOK   8192
#define DM      1024
#define DI      2048
#define NH      32
#define HD      64
#define DS      16
#define CDIM    2080
#define RAWN    2112      // in_proj cols [2048,4160): xBC(2080) + dt(32)
#define RAWNP   2176      // RAWN padded to a multiple of 128 (zero-filled)
#define IPLD    4160      // in_proj row stride

typedef __attribute__((ext_vector_type(16))) __bf16 v16bf;
typedef __attribute__((ext_vector_type(8)))  float  v8f;
typedef unsigned short ushort_t;

static __device__ __forceinline__ unsigned bf16bits(float f) {
  unsigned u = __float_as_uint(f);
  return ((u + 0x7FFFu + ((u >> 16) & 1u)) >> 16) & 0xFFFFu;   // RNE
}
static __device__ __forceinline__ float bf2f(ushort_t s) {
  return __uint_as_float(((unsigned)s) << 16);
}
static __device__ __forceinline__ float siluf(float x) {
  return x / (1.f + __expf(-x));
}
static __device__ __forceinline__ float gelu_exact(float x) {
  return 0.5f * x * (1.f + erff(x * 0.70710678118654752440f));
}

// ---- gfx1250 async global->LDS copy (probe-guarded, sync fallback) --------
#if defined(__has_builtin)
#if __has_builtin(__builtin_amdgcn_global_load_async_to_lds_b128)
#define HAVE_ASYNC_LDS 1
#endif
#endif
#ifndef HAVE_ASYNC_LDS
#define HAVE_ASYNC_LDS 0
#endif

static __device__ __forceinline__ void copy16_g2l(const ushort_t* g, ushort_t* l) {
#if HAVE_ASYNC_LDS
  typedef int v4i_t __attribute__((vector_size(4 * sizeof(int))));
  __builtin_amdgcn_global_load_async_to_lds_b128(
      (__attribute__((address_space(1))) v4i_t*)((void*)g),
      (__attribute__((address_space(3))) v4i_t*)((void*)l), 0, 0);
#else
  *(uint4*)l = *(const uint4*)g;
#endif
}
static __device__ __forceinline__ void wait_async_lds() {
#if HAVE_ASYNC_LDS
#if __has_builtin(__builtin_amdgcn_s_wait_asynccnt)
  __builtin_amdgcn_s_wait_asynccnt(0);
#else
  asm volatile("s_wait_asynccnt 0x0" ::: "memory");
#endif
#endif
}

// ---------------------------------------------------------------------------
// WMMA bf16 GEMM: C[M x N](f32) = A[M x K](bf16) @ B[K x N](bf16).
// Tile 128x128x32, 256 threads (8 waves, 4x2), wave -> 32x64 (2x4 WMMA).
// Ping-pong LDS buffers; tile k+1 is DMA'd while tile k is multiplied.
// Loads are guard-free (M%128==0, K%32==0, B padded to gridDim.x*128 cols);
// only the C store is guarded by N.
// ---------------------------------------------------------------------------
__global__ __launch_bounds__(256) void gemm_bf16_wmma(
    const ushort_t* __restrict__ A, const ushort_t* __restrict__ B,
    float* __restrict__ C, int N, int K, int lda, int ldb, int ldc) {
  const int tid  = threadIdx.x;
  const int lane = tid & 31;
  const int wid  = tid >> 5;
  const int wn   = wid & 1;        // 0..1 -> 64-col half
  const int wm   = wid >> 1;       // 0..3 -> 32-row quarter
  const int bm0  = blockIdx.y * 128;
  const int bn0  = blockIdx.x * 128;

  // row strides chosen so every 16B chunk lands 16B-aligned in LDS
  __shared__ ushort_t As[2][128][40];   // 32 data + 8 pad halves (80 B rows)
  __shared__ ushort_t Bs[2][32][136];   // 128 data + 8 pad halves (272 B rows)

#define STAGE_TILE(KT, BUF)                                                   \
  {                                                                           \
    _Pragma("unroll")                                                         \
    for (int r_ = 0; r_ < 2; ++r_) {                                          \
      int idx_ = tid + 256 * r_;                                              \
      int row_ = idx_ >> 2, c_ = (idx_ & 3) * 8;                              \
      copy16_g2l(A + (size_t)(bm0 + row_) * lda + (KT)*32 + c_,               \
                 &As[BUF][row_][c_]);                                         \
    }                                                                         \
    _Pragma("unroll")                                                         \
    for (int r_ = 0; r_ < 2; ++r_) {                                          \
      int idx_ = tid + 256 * r_;                                              \
      int row_ = idx_ >> 4, c_ = (idx_ & 15) * 8;                             \
      copy16_g2l(B + (size_t)((KT)*32 + row_) * ldb + bn0 + c_,               \
                 &Bs[BUF][row_][c_]);                                         \
    }                                                                         \
  }

  v8f acc[2][4];
#pragma unroll
  for (int i = 0; i < 2; ++i)
#pragma unroll
    for (int j = 0; j < 4; ++j)
#pragma unroll
      for (int v = 0; v < 8; ++v) acc[i][j][v] = 0.f;

  const int mrow = lane & 15;
  const int half = lane >> 4;
  const int nk = K >> 5;

  STAGE_TILE(0, 0);
  wait_async_lds();
  __syncthreads();

  for (int kt = 0; kt < nk; ++kt) {
    const int cur = kt & 1;
    if (kt + 1 < nk) STAGE_TILE(kt + 1, cur ^ 1);   // overlap with WMMA below

    v16bf afr[2], bfr[4];
#pragma unroll
    for (int i = 0; i < 2; ++i) {
      int m = wm * 32 + i * 16 + mrow;
      union { unsigned u[8]; v16bf v; } t;
#pragma unroll
      for (int vv = 0; vv < 8; ++vv) {
        int kk = ((vv < 4) ? (2 * vv) : (16 + 2 * (vv - 4))) + 8 * half;
        t.u[vv] = *(const unsigned*)&As[cur][m][kk];   // K-pair (kk, kk+1)
      }
      afr[i] = t.v;
    }
#pragma unroll
    for (int j = 0; j < 4; ++j) {
      int n0 = wn * 64 + j * 16;
      union { ushort_t s[16]; v16bf v; } t;
#pragma unroll
      for (int s = 0; s < 16; ++s) {
        int kk = ((s < 8) ? s : (8 + s)) + 8 * half;
        t.s[s] = Bs[cur][kk][n0 + mrow];
      }
      bfr[j] = t.v;
    }
#pragma unroll
    for (int i = 0; i < 2; ++i)
#pragma unroll
      for (int j = 0; j < 4; ++j)
        acc[i][j] = __builtin_amdgcn_wmma_f32_16x16x32_bf16(
            false, afr[i], false, bfr[j], (short)0, acc[i][j], false, false);

    wait_async_lds();     // next tile's DMA has had the whole WMMA block
    __syncthreads();
  }
#undef STAGE_TILE

  // C layout: row = v + 8*half, col = lane&15
#pragma unroll
  for (int i = 0; i < 2; ++i)
#pragma unroll
    for (int j = 0; j < 4; ++j) {
      int mbase = bm0 + wm * 32 + i * 16 + half * 8;
      int n     = bn0 + wn * 64 + j * 16 + mrow;
      if (n < N) {
#pragma unroll
        for (int vv = 0; vv < 8; ++vv)
          C[(size_t)(mbase + vv) * ldc + n] = acc[i][j][vv];
      }
    }
}

// ---------------------------------------------------------------------------
// One-time f32 -> bf16 weight conversion, with optional column slice + pad.
// ---------------------------------------------------------------------------
__global__ __launch_bounds__(256) void cvt_w_kernel(
    const float* __restrict__ src, ushort_t* __restrict__ dst,
    int srcld, int srcoff, int width, int dstld, int total) {
  int idx = blockIdx.x * 256 + threadIdx.x;
  if (idx >= total) return;
  int r = idx / dstld, c = idx - r * dstld;
  dst[idx] = (c < width) ? (ushort_t)bf16bits(src[(size_t)r * srcld + srcoff + c])
                         : (ushort_t)0;
}

// ---------------------------------------------------------------------------
// Layer1: (T x 64) @ (64 x 1024) + bias -> LN -> GELU -> bf16, block/token.
// ---------------------------------------------------------------------------
__global__ __launch_bounds__(256) void embed_l1_kernel(
    const float* __restrict__ x, const float* __restrict__ w1,
    const float* __restrict__ b1, const float* __restrict__ g,
    const float* __restrict__ bv, ushort_t* __restrict__ outbf) {
  const int t = blockIdx.x, tid = threadIdx.x;
  __shared__ float xs[64];
  __shared__ float red[256];
  if (tid < 64) xs[tid] = x[(size_t)t * 64 + tid];
  __syncthreads();

  float v[4], s = 0.f, ss = 0.f;
#pragma unroll
  for (int r = 0; r < 4; ++r) {
    int j = tid + 256 * r;
    float a = b1[j];
    for (int k = 0; k < 64; ++k) a += xs[k] * w1[k * DM + j];
    v[r] = a; s += a; ss += a * a;
  }
  red[tid] = s; __syncthreads();
  for (int st = 128; st > 0; st >>= 1) { if (tid < st) red[tid] += red[tid + st]; __syncthreads(); }
  float mean = red[0] * (1.f / DM); __syncthreads();
  red[tid] = ss; __syncthreads();
  for (int st = 128; st > 0; st >>= 1) { if (tid < st) red[tid] += red[tid + st]; __syncthreads(); }
  float var = red[0] * (1.f / DM) - mean * mean;
  float inv = rsqrtf(var + 1e-5f);
#pragma unroll
  for (int r = 0; r < 4; ++r) {
    int j = tid + 256 * r;
    float a = (v[r] - mean) * inv * g[j] + bv[j];
    outbf[(size_t)t * DM + j] = (ushort_t)bf16bits(gelu_exact(a));
  }
}

// ---------------------------------------------------------------------------
// in(f32) + bias -> LN(g,b) -> optional GELU -> out bf16. Block per token.
// ---------------------------------------------------------------------------
__global__ __launch_bounds__(256) void ln_act_kernel(
    const float* __restrict__ in, ushort_t* __restrict__ outbf,
    const float* __restrict__ bias, const float* __restrict__ g,
    const float* __restrict__ bv, int do_gelu) {
  const int t = blockIdx.x, tid = threadIdx.x;
  __shared__ float red[256];
  float v[4], s = 0.f, ss = 0.f;
#pragma unroll
  for (int r = 0; r < 4; ++r) {
    int j = tid + 256 * r;
    float a = in[(size_t)t * DM + j] + bias[j];
    v[r] = a; s += a; ss += a * a;
  }
  red[tid] = s; __syncthreads();
  for (int st = 128; st > 0; st >>= 1) { if (tid < st) red[tid] += red[tid + st]; __syncthreads(); }
  float mean = red[0] * (1.f / DM); __syncthreads();
  red[tid] = ss; __syncthreads();
  for (int st = 128; st > 0; st >>= 1) { if (tid < st) red[tid] += red[tid + st]; __syncthreads(); }
  float var = red[0] * (1.f / DM) - mean * mean;
  float inv = rsqrtf(var + 1e-5f);
#pragma unroll
  for (int r = 0; r < 4; ++r) {
    int j = tid + 256 * r;
    float a = (v[r] - mean) * inv * g[j] + bv[j];
    outbf[(size_t)t * DM + j] = (ushort_t)bf16bits(do_gelu ? gelu_exact(a) : a);
  }
}

// z (in_proj cols 0..2047) for the 8 last tokens only: 16384 outputs.
__global__ __launch_bounds__(256) void z_last_kernel(
    const ushort_t* __restrict__ h3bf, const float* __restrict__ W,
    float* __restrict__ z_last) {
  int idx = blockIdx.x * 256 + threadIdx.x;       // 0..16383
  int b = idx >> 11, j = idx & 2047;
  const ushort_t* hr = h3bf + (size_t)(b * 1024 + 1023) * DM;
  float a = 0.f;
  for (int k = 0; k < DM; ++k) a += bf2f(hr[k]) * W[(size_t)k * IPLD + j];
  z_last[idx] = a;
}

// dt = softplus(raw_dt + dt_bias); decay = exp(dt * (-exp(A_log)))
__global__ __launch_bounds__(256) void dt_decay_kernel(
    const float* __restrict__ raw, const float* __restrict__ dt_bias,
    const float* __restrict__ A_log, float* __restrict__ dtv,
    float* __restrict__ decv) {
  int idx = blockIdx.x * 256 + threadIdx.x;       // T*32
  int t = idx >> 5, h = idx & 31;
  float xv = raw[(size_t)t * RAWN + CDIM + h] + dt_bias[h];
  float sp = (xv > 20.f) ? xv : log1pf(__expf(xv));
  dtv[idx]  = sp;
  decv[idx] = __expf(-__expf(A_log[h]) * sp);
}

// causal depthwise conv(4) + SiLU over 2080 channels, per (token, channel)
__global__ __launch_bounds__(256) void conv_silu_kernel(
    const float* __restrict__ raw, const float* __restrict__ cw,
    const float* __restrict__ cb, float* __restrict__ xconv) {
  int t = blockIdx.x;
  int c = blockIdx.y * 256 + threadIdx.x;
  if (c >= CDIM) return;
  int l = t & 1023;
  float a = cb[c];
#pragma unroll
  for (int k = 0; k < 4; ++k) {
    int ll = l - 3 + k;
    if (ll >= 0) a += cw[k * CDIM + c] * raw[(size_t)(t - 3 + k) * RAWN + c];
  }
  xconv[(size_t)t * CDIM + c] = siluf(a);
}

// SSM scan: block per (b,h); thread owns (p = tid>>2, n = (tid&3)*4 .. +3).
// Only y at t = L-1 is needed downstream.
__global__ __launch_bounds__(256) void scan_kernel(
    const float* __restrict__ xconv, const float* __restrict__ dtv,
    const float* __restrict__ decv, const float* __restrict__ D_skip,
    float* __restrict__ y_last) {
  const int bh = blockIdx.x, b = bh >> 5, h = bh & 31;
  const int tid = threadIdx.x;
  const int p = tid >> 2, nb = (tid & 3) * 4;
  __shared__ float sx[64], sB[16], sC[16], sS[2];
  __shared__ float yred[256];
  float S0 = 0.f, S1 = 0.f, S2 = 0.f, S3 = 0.f;
  const float dsk = D_skip[h];

  for (int l = 0; l < 1024; ++l) {
    int t = b * 1024 + l;
    const float* row = xconv + (size_t)t * CDIM;
    if (tid < 64)       sx[tid]      = row[h * HD + tid];
    else if (tid < 80)  sB[tid - 64] = row[DI + (tid - 64)];
    else if (tid < 96)  sC[tid - 80] = row[DI + DS + (tid - 80)];
    else if (tid == 96) sS[0] = decv[t * NH + h];
    else if (tid == 97) sS[1] = dtv[t * NH + h];
    __syncthreads();
    float dec = sS[0];
    float w   = sS[1] * sx[p];
    S0 = dec * S0 + w * sB[nb + 0];
    S1 = dec * S1 + w * sB[nb + 1];
    S2 = dec * S2 + w * sB[nb + 2];
    S3 = dec * S3 + w * sB[nb + 3];
    __syncthreads();
  }
  yred[tid] = S0 * sC[nb + 0] + S1 * sC[nb + 1] + S2 * sC[nb + 2] + S3 * sC[nb + 3];
  __syncthreads();
  if ((tid & 3) == 0) {
    float y = yred[tid] + yred[tid + 1] + yred[tid + 2] + yred[tid + 3] + dsk * sx[p];
    y_last[(size_t)b * DI + h * HD + p] = y;
  }
}

// gate with silu(z), RMSNorm over 2048, scale by norm_w. Block per batch.
__global__ __launch_bounds__(256) void gate_rms_kernel(
    const float* __restrict__ y_last, const float* __restrict__ z_last,
    const float* __restrict__ norm_w, float* __restrict__ yn) {
  const int b = blockIdx.x, tid = threadIdx.x;
  __shared__ float red[256];
  float v[8], ss = 0.f;
#pragma unroll
  for (int r = 0; r < 8; ++r) {
    int j = tid + 256 * r;
    float y = y_last[(size_t)b * DI + j] * siluf(z_last[(size_t)b * DI + j]);
    v[r] = y; ss += y * y;
  }
  red[tid] = ss; __syncthreads();
  for (int st = 128; st > 0; st >>= 1) { if (tid < st) red[tid] += red[tid + st]; __syncthreads(); }
  float inv = rsqrtf(red[0] * (1.f / DI) + 1e-5f);
#pragma unroll
  for (int r = 0; r < 8; ++r) {
    int j = tid + 256 * r;
    yn[(size_t)b * DI + j] = v[r] * inv * norm_w[j];
  }
}

// out_proj on last tokens only: (8 x 2048) @ (2048 x 1024)
__global__ __launch_bounds__(256) void outproj_last_kernel(
    const float* __restrict__ yn, const float* __restrict__ W,
    float* __restrict__ o_last) {
  int idx = blockIdx.x * 256 + threadIdx.x;       // 0..8191
  int b = idx >> 10, j = idx & 1023;
  float a = 0.f;
  for (int k = 0; k < DI; ++k) a += yn[(size_t)b * DI + k] * W[(size_t)k * DM + j];
  o_last[idx] = a;
}

// classifier: (8 x 1024) @ (1024 x 64) + bo -> 512 outputs
__global__ __launch_bounds__(256) void final_kernel(
    const float* __restrict__ o_last, const float* __restrict__ wo,
    const float* __restrict__ bo, float* __restrict__ out) {
  int idx = blockIdx.x * 256 + threadIdx.x;       // 0..511
  if (idx >= 512) return;
  int b = idx >> 6, j = idx & 63;
  float a = bo[j];
  for (int k = 0; k < DM; ++k) a += o_last[b * DM + k] * wo[k * 64 + j];
  out[idx] = a;
}

// ---------------------------------------------------------------------------
extern "C" void kernel_launch(void* const* d_in, const int* in_sizes, int n_in,
                              void* d_out, int out_size, void* d_ws, size_t ws_size,
                              hipStream_t stream) {
  const float* x        = (const float*)d_in[0];
  const float* w_in1    = (const float*)d_in[1];
  const float* b_in1    = (const float*)d_in[2];
  const float* ln1_g    = (const float*)d_in[3];
  const float* ln1_b    = (const float*)d_in[4];
  const float* w_in2    = (const float*)d_in[5];
  const float* b_in2    = (const float*)d_in[6];
  const float* ln2_g    = (const float*)d_in[7];
  const float* ln2_b    = (const float*)d_in[8];
  const float* w_in3    = (const float*)d_in[9];
  const float* b_in3    = (const float*)d_in[10];
  const float* ln3_g    = (const float*)d_in[11];
  const float* ln3_b    = (const float*)d_in[12];
  const float* in_proj  = (const float*)d_in[13];
  const float* conv_w   = (const float*)d_in[14];
  const float* conv_b   = (const float*)d_in[15];
  const float* dt_bias  = (const float*)d_in[16];
  const float* A_log    = (const float*)d_in[17];
  const float* D_skip   = (const float*)d_in[18];
  const float* norm_w   = (const float*)d_in[19];
  const float* out_proj = (const float*)d_in[20];
  const float* wo       = (const float*)d_in[21];
  const float* bo       = (const float*)d_in[22];
  float* out = (float*)d_out;

  // ---- workspace carve-up -------------------------------------------------
  char* p = (char*)d_ws;
  float* gout   = (float*)p;  p += (size_t)T_TOK * DM * 4;        // GEMM2/3 f32 out
  float* raw    = (float*)p;  p += (size_t)T_TOK * RAWN * 4;      // in_proj f32 out
  float* xconv  = (float*)p;  p += (size_t)T_TOK * CDIM * 4;
  float* dtv    = (float*)p;  p += (size_t)T_TOK * NH * 4;
  float* decv   = (float*)p;  p += (size_t)T_TOK * NH * 4;
  float* z_last = (float*)p;  p += 8 * DI * 4;
  float* y_last = (float*)p;  p += 8 * DI * 4;
  float* yn     = (float*)p;  p += 8 * DI * 4;
  float* o_last = (float*)p;  p += 8 * DM * 4;
  ushort_t* hAbf = (ushort_t*)p; p += (size_t)T_TOK * DM * 2;     // act bf16 (A/C ping)
  ushort_t* hBbf = (ushort_t*)p; p += (size_t)T_TOK * DM * 2;     // act bf16 (B ping)
  ushort_t* wB2  = (ushort_t*)p; p += (size_t)DM * DM * 2;
  ushort_t* wB3  = (ushort_t*)p; p += (size_t)DM * DM * 2;
  ushort_t* wIP  = (ushort_t*)p; p += (size_t)DM * RAWNP * 2;     // padded slice

  // ---- one-time weight bf16 conversion ------------------------------------
  cvt_w_kernel<<<(DM * DM + 255) / 256, 256, 0, stream>>>(
      w_in2, wB2, DM, 0, DM, DM, DM * DM);
  cvt_w_kernel<<<(DM * DM + 255) / 256, 256, 0, stream>>>(
      w_in3, wB3, DM, 0, DM, DM, DM * DM);
  cvt_w_kernel<<<(DM * RAWNP + 255) / 256, 256, 0, stream>>>(
      in_proj, wIP, IPLD, DI, RAWN, RAWNP, DM * RAWNP);

  // ---- layer 1 (fused GEMM + LN + GELU -> bf16) ---------------------------
  embed_l1_kernel<<<T_TOK, 256, 0, stream>>>(x, w_in1, b_in1, ln1_g, ln1_b, hAbf);
  // ---- layer 2 ------------------------------------------------------------
  gemm_bf16_wmma<<<dim3(DM / 128, T_TOK / 128), 256, 0, stream>>>(
      hAbf, wB2, gout, DM, DM, DM, DM, DM);
  ln_act_kernel<<<T_TOK, 256, 0, stream>>>(gout, hBbf, b_in2, ln2_g, ln2_b, 1);
  // ---- layer 3 ------------------------------------------------------------
  gemm_bf16_wmma<<<dim3(DM / 128, T_TOK / 128), 256, 0, stream>>>(
      hBbf, wB3, gout, DM, DM, DM, DM, DM);
  ln_act_kernel<<<T_TOK, 256, 0, stream>>>(gout, hAbf, b_in3, ln3_g, ln3_b, 0);
  // ---- in_proj: xBC+dt columns for all tokens; z only at last tokens ------
  gemm_bf16_wmma<<<dim3(RAWNP / 128, T_TOK / 128), 256, 0, stream>>>(
      hAbf, wIP, raw, RAWN, DM, DM, RAWNP, RAWN);
  z_last_kernel<<<64, 256, 0, stream>>>(hAbf, in_proj, z_last);
  // ---- dt/decay, conv+silu ------------------------------------------------
  dt_decay_kernel<<<(T_TOK * NH) / 256, 256, 0, stream>>>(raw, dt_bias, A_log, dtv, decv);
  conv_silu_kernel<<<dim3(T_TOK, (CDIM + 255) / 256), 256, 0, stream>>>(raw, conv_w, conv_b, xconv);
  // ---- SSM scan (one block per (b,h)) -------------------------------------
  scan_kernel<<<8 * NH, 256, 0, stream>>>(xconv, dtv, decv, D_skip, y_last);
  // ---- tail on last tokens only -------------------------------------------
  gate_rms_kernel<<<8, 256, 0, stream>>>(y_last, z_last, norm_w, yn);
  outproj_last_kernel<<<(8 * DM) / 256, 256, 0, stream>>>(yn, out_proj, o_last);
  final_kernel<<<2, 256, 0, stream>>>(o_last, wo, bo, out);
}
